// P4Transformer_17686675325595
// MI455X (gfx1250) — matrix operand order repaired and checked by reading, not compiled, not run on hardware
//
#include <hip/hip_runtime.h>
#include <math.h>

// ---------------------------------------------------------------------------
// CDNA5 (gfx1250) P4Transformer forward.
// Single GEMM engine: all operands pre-packed f16 (K-major), WMMA
// v_wmma_f32_16x16x32_f16, wave32, 8 waves/block, 128x128 block tile,
// 32x64 per-wave tile (8 WMMA / K-step), double-buffered LDS staged by the
// Tensor Data Mover (tensor_load_to_lds + s_wait_tensorcnt) with the D#
// doing OOB zero-fill and LDS row padding in hardware. Fused epilogue:
// alpha * acc * gamma[col] + beta[col] + rbeta[row] -> relu/gelu -> f32/f16,
// optional accumulate and transposed store.
// ---------------------------------------------------------------------------

typedef _Float16 v16h __attribute__((ext_vector_type(16)));
typedef _Float16 h8   __attribute__((ext_vector_type(8)));
typedef float    v8f  __attribute__((ext_vector_type(8)));
typedef unsigned int u32x4 __attribute__((ext_vector_type(4)));
typedef int          i32x4v __attribute__((ext_vector_type(4)));
typedef int          i32x8v __attribute__((ext_vector_type(8)));

#if __has_builtin(__builtin_amdgcn_tensor_load_to_lds) && __has_builtin(__builtin_amdgcn_s_wait_tensorcnt)
#define USE_TDM 1
#else
#define USE_TDM 0
#endif

__device__ __forceinline__ v16h frag16(h8 lo, h8 hi) {
    return __builtin_shufflevector(lo, hi, 0,1,2,3,4,5,6,7,8,9,10,11,12,13,14,15);
}

#define GBM 128
#define GBN 128
#define GBK 32
#define LDSK 40   // padded LDS row stride in halfs (80B, 16B aligned, conflict-free)

#if USE_TDM
// One TDM load: (128 x 32) f16 tile from a K-major matrix into LDS.
// pad_interval = 16 DWORDs (32 halfs), pad_amount = 4 DWORDs (8 halfs) -> 40-half rows.
// tensor_dim0/1 = remaining valid extent; TDM zero-fills OOB rows/cols.
__device__ __forceinline__ void tdm_tile_f16(const _Float16* gbase, long ld,
                                             unsigned ldsByteAddr,
                                             unsigned tdim0, unsigned tdim1)
{
    unsigned long long ga = (unsigned long long)(__SIZE_TYPE__)gbase;
    u32x4 g0;
    g0.x = 1u;                                                // count=1
    g0.y = ldsByteAddr;                                       // lds_addr
    g0.z = (unsigned)ga;                                      // global_addr lo
    g0.w = (unsigned)((ga >> 32) & 0x1FFFFFFu) | (2u << 30);  // addr[56:32] | type=2
    i32x8v g1;
    g1[0] = (int)((1u << 16) | (1u << 20) | (3u << 22) | (3u << 25));
    g1[1] = (int)((tdim0 & 0xFFFFu) << 16);
    g1[2] = (int)((tdim0 >> 16) | ((tdim1 & 0xFFFFu) << 16));
    g1[3] = (int)((tdim1 >> 16) | (32u << 16));               // tile_dim0 = 32
    g1[4] = (int)128;                                         // tile_dim1 = 128
    g1[5] = (int)(unsigned)ld;                                // dim0 stride (halfs)
    g1[6] = 0;
    g1[7] = 0;
    i32x4v z4 = {0, 0, 0, 0};
#if __clang_major__ >= 23
    i32x8v z8 = {0, 0, 0, 0, 0, 0, 0, 0};
    __builtin_amdgcn_tensor_load_to_lds(g0, g1, z4, z4, z8, 0);
#else
    __builtin_amdgcn_tensor_load_to_lds(g0, g1, z4, z4, 0);
#endif
}
#endif

// C[M,N] = act( alpha*(A x B^T)*gamma[col] + beta[col] + rbeta[row] ) (+C)
// A: MxK f16 K-major (lda in halfs, rows need not be padded; K-pad handled)
// B: NxK f16 K-major (weight layout)
__launch_bounds__(256)
__global__ void k_gemm_wmma(const _Float16* __restrict__ A, long lda,
                            const _Float16* __restrict__ Bm, long ldb,
                            void* __restrict__ Cv, long ldc,
                            int M, int N, int K, float alpha,
                            const float* __restrict__ gamma,
                            const float* __restrict__ beta,
                            const float* __restrict__ rbeta,
                            int act, int accum, int outF16, int outTrans)
{
    __shared__ _Float16 sA[2 * GBM * LDSK];
    __shared__ _Float16 sB[2 * GBN * LDSK];

    const int tid  = threadIdx.x;
    const int lane = tid & 31;
    const int wave = tid >> 5;
    const int wm   = wave & 3;     // 4 wave-rows x 32
    const int wn   = wave >> 2;    // 2 wave-cols x 64
    const int bm0  = blockIdx.y * GBM;
    const int bn0  = blockIdx.x * GBN;
    const int KT   = (K + GBK - 1) / GBK;

    v8f acc[2][4] = {};

    auto stage_tile = [&](int which, int buf, int k0) {
        const _Float16* src = which ? Bm : A;
        long ld = which ? ldb : lda;
        int  V  = which ? N : M;
        int  v0 = which ? bn0 : bm0;
        _Float16* sbase = which ? &sB[buf * GBN * LDSK] : &sA[buf * GBM * LDSK];
#if USE_TDM
        if (wave == 0)
            tdm_tile_f16(src + (long)v0 * ld + k0, ld,
                         (unsigned)(__SIZE_TYPE__)sbase,
                         (unsigned)(K - k0), (unsigned)(V - v0));
#else
        for (int it = 0; it < 2; ++it) {           // 128 rows x 4 b128-chunks
            int idx = tid + it * 256;
            int r = idx >> 2, c8 = (idx & 3) << 3;
            long gr = v0 + r; if (gr >= V) gr = V - 1;
            const _Float16* sp = src + gr * ld + k0 + c8;
            _Float16* dst = sbase + r * LDSK + c8;
            if (k0 + c8 + 8 <= K) *(h8*)dst = *(const h8*)sp;
            else for (int j = 0; j < 8; ++j) dst[j] = (k0 + c8 + j < K) ? sp[j] : (_Float16)0.f;
        }
#endif
    };

    stage_tile(0, 0, 0);
    stage_tile(1, 0, 0);
#if USE_TDM
    if (wave == 0) __builtin_amdgcn_s_wait_tensorcnt(0);
#endif
    __syncthreads();

    int buf = 0;
    const int hsel = lane >> 4;
    const int msel = lane & 15;

    for (int kt = 0; kt < KT; ++kt) {
        if (kt + 1 < KT) { stage_tile(0, buf ^ 1, (kt + 1) * GBK); stage_tile(1, buf ^ 1, (kt + 1) * GBK); }

        v16h afr[2], bfr[4];
        for (int i = 0; i < 2; ++i) {
            // A 16x32 fragment: lane = row m, halfs [h*8,h*8+8) U [h*8+16,h*8+24)
            const _Float16* pa = &sA[buf * GBM * LDSK + (wm * 32 + i * 16 + msel) * LDSK + hsel * 8];
            afr[i] = frag16(*(const h8*)pa, *(const h8*)(pa + 16));
        }
        for (int j = 0; j < 4; ++j) {
            // B 32x16 fragment: lane = col n, halfs [h*8,h*8+8) U [h*8+16,h*8+24)
            const _Float16* pb = &sB[buf * GBN * LDSK + (wn * 64 + j * 16 + msel) * LDSK + hsel * 8];
            bfr[j] = frag16(*(const h8*)pb, *(const h8*)(pb + 16));
        }
        for (int i = 0; i < 2; ++i)
            for (int j = 0; j < 4; ++j)
                acc[i][j] = __builtin_amdgcn_wmma_f32_16x16x32_f16(
                    false, afr[i], false, bfr[j], (short)0, acc[i][j], false, false);

#if USE_TDM
        if (wave == 0) __builtin_amdgcn_s_wait_tensorcnt(0);
#endif
        __syncthreads();
        buf ^= 1;
    }

    for (int i = 0; i < 2; ++i)
        for (int j = 0; j < 4; ++j) {
            int col   = bn0 + wn * 64 + j * 16 + msel;
            int rbase = bm0 + wm * 32 + i * 16 + (hsel << 3);
            if (col >= N) continue;
            float g  = gamma ? gamma[col] : 1.f;
            float bb = beta  ? beta[col]  : 0.f;
            for (int r = 0; r < 8; ++r) {
                int row = rbase + r;
                if (row >= M) continue;
                float v = alpha * acc[i][j][r] * g + bb;
                if (rbeta) v += rbeta[row];
                if (act == 1) v = fmaxf(v, 0.f);
                else if (act == 2) v = 0.5f * v * (1.f + erff(v * 0.70710678f));
                long idx = outTrans ? ((long)col * ldc + row) : ((long)row * ldc + col);
                if (outF16) {
                    _Float16* pc = (_Float16*)Cv + idx;
                    *pc = accum ? (_Float16)((float)*pc + v) : (_Float16)v;
                } else {
                    float* pc = (float*)Cv + idx;
                    *pc = accum ? (*pc + v) : v;
                }
            }
        }
}

// ------------------------- auxiliary kernels -------------------------------

// f32 -> f16 pack, optional transpose, K zero-padded to KP columns.
__global__ void k_packh(const float* __restrict__ src, long ld, int trans,
                        _Float16* __restrict__ dst, long M, long K, long KP)
{
    long i = blockIdx.x * (long)blockDim.x + threadIdx.x;
    if (i >= M * KP) return;
    long k = i % KP, m = i / KP;
    float v = (k < K) ? (trans ? src[k * ld + m] : src[m * ld + k]) : 0.f;
    dst[i] = (_Float16)v;
}

// f16 -> f16 pack (transpose), for attention V^T from packed qkv.
__global__ void k_packh16(const _Float16* __restrict__ src, long ld, int trans,
                          _Float16* __restrict__ dst, long M, long K)
{
    long i = blockIdx.x * (long)blockDim.x + threadIdx.x;
    if (i >= M * K) return;
    long k = i % K, m = i / K;
    dst[i] = trans ? src[k * ld + m] : src[m * ld + k];
}

__global__ void k_fps(const float* __restrict__ xyz, long bs, int N, int M,
                      int* __restrict__ idx, long idx_bs, float* __restrict__ distws)
{
    const float* P = xyz + (long)blockIdx.x * bs;
    int* out = idx + (long)blockIdx.x * idx_bs;
    float* dist = distws + (long)blockIdx.x * N;
    __shared__ float sval[256];
    __shared__ int   sidx[256];
    __shared__ int   s_last;
    int tid = threadIdx.x;
    for (int n = tid; n < N; n += 256) dist[n] = 1e10f;
    if (tid == 0) { out[0] = 0; s_last = 0; }
    __syncthreads();
    for (int it = 1; it < M; ++it) {
        int last = s_last;
        float lx = P[3*last], ly = P[3*last+1], lz = P[3*last+2];
        float best = -1.f; int bi = 0;
        for (int n = tid; n < N; n += 256) {
            float dx = P[3*n]-lx, dy = P[3*n+1]-ly, dz = P[3*n+2]-lz;
            float dn = fminf(dist[n], dx*dx + dy*dy + dz*dz);
            dist[n] = dn;
            if (dn > best) { best = dn; bi = n; }
        }
        sval[tid] = best; sidx[tid] = bi;
        __syncthreads();
        for (int s = 128; s > 0; s >>= 1) {
            if (tid < s && sval[tid+s] > sval[tid]) { sval[tid]=sval[tid+s]; sidx[tid]=sidx[tid+s]; }
            __syncthreads();
        }
        if (tid == 0) { out[it] = sidx[0]; s_last = sidx[0]; }
        __syncthreads();
    }
}

__global__ void k_gather_rows(const float* __restrict__ src, long src_bs,
                              const int* __restrict__ idx, long idx_bs,
                              float* __restrict__ dst, long dst_bs,
                              int M, int C, int nB)
{
    long i = blockIdx.x * (long)blockDim.x + threadIdx.x;
    long total = (long)nB * M * C;
    if (i >= total) return;
    int c = (int)(i % C); long t = i / C;
    int m = (int)(t % M); int b = (int)(t / M);
    int r = idx[(long)b * idx_bs + m];
    dst[(long)b * dst_bs + (long)m * C + c] = src[(long)b * src_bs + (long)r * C + c];
}

// feature gather, writes f16 GEMM operand directly
__global__ void k_gather_rows_h(const float* __restrict__ src, long src_bs,
                                const int* __restrict__ idx, long idx_bs,
                                _Float16* __restrict__ dst, long dst_bs,
                                int M, int C, int nB)
{
    long i = blockIdx.x * (long)blockDim.x + threadIdx.x;
    long total = (long)nB * M * C;
    if (i >= total) return;
    int c = (int)(i % C); long t = i / C;
    int m = (int)(t % M); int b = (int)(t / M);
    int r = idx[(long)b * idx_bs + m];
    dst[(long)b * dst_bs + (long)m * C + c] = (_Float16)src[(long)b * src_bs + (long)r * C + c];
}

__global__ void k_ballq(const float* __restrict__ xyz, long xyz_bs,
                        const float* __restrict__ anchors, long anc_bs,
                        int* __restrict__ out, int N, int M, int k, float r2, int nB)
{
    long i = blockIdx.x * (long)blockDim.x + threadIdx.x;
    if (i >= (long)nB * M) return;
    int b = (int)(i / M), m = (int)(i % M);
    const float* P = xyz + (long)b * xyz_bs;
    const float* a = anchors + (long)b * anc_bs + 3L * m;
    float ax = a[0], ay = a[1], az = a[2];
    int* o = out + i * (long)k;
    int cnt = 0, first = -1;
    for (int n = 0; n < N && cnt < k; ++n) {
        float dx = P[3*n]-ax, dy = P[3*n+1]-ay, dz = P[3*n+2]-az;
        if (dx*dx + dy*dy + dz*dz < r2) { if (first < 0) first = n; o[cnt++] = n; }
    }
    int fill = first >= 0 ? first : 0;
    for (; cnt < k; ++cnt) o[cnt] = fill;
}

// displacement + time feature (f16 GEMM operand, K=4)
__global__ void k_group_disp(const float* __restrict__ xyz, long xyz_bs,
                             const float* __restrict__ anchors, long anc_bs,
                             const int* __restrict__ nidx,
                             _Float16* __restrict__ out, int M, int k, float td, int nB)
{
    long i = blockIdx.x * (long)blockDim.x + threadIdx.x;
    long total = (long)nB * M * k;
    if (i >= total) return;
    long t = i / k;
    int m = (int)(t % M); int b = (int)(t / M);
    int n = nidx[i];
    const float* p = xyz + (long)b * xyz_bs + 3L * n;
    const float* a = anchors + (long)b * anc_bs + 3L * m;
    _Float16* o = out + i * 4;
    o[0] = (_Float16)(p[0]-a[0]); o[1] = (_Float16)(p[1]-a[1]);
    o[2] = (_Float16)(p[2]-a[2]); o[3] = (_Float16)td;
}

// a + b -> f16 (fused add + pack between CBR stages)
__global__ void k_addh(const float* __restrict__ a, const float* __restrict__ b,
                       _Float16* __restrict__ c, long n)
{
    long i = blockIdx.x * (long)blockDim.x + threadIdx.x;
    if (i < n) c[i] = (_Float16)(a[i] + b[i]);
}

__global__ void k_relu(float* __restrict__ p, long n)
{
    long i = blockIdx.x * (long)blockDim.x + threadIdx.x;
    if (i < n) p[i] = fmaxf(p[i], 0.f);
}

__global__ void k_fill(float* __restrict__ p, long n, float v)
{
    long i = blockIdx.x * (long)blockDim.x + threadIdx.x;
    if (i < n) p[i] = v;
}

__global__ void k_maxpool(const float* __restrict__ in, float* __restrict__ out,
                          long out_bs, int M, int k, int C, int nB, int accum)
{
    long i = blockIdx.x * (long)blockDim.x + threadIdx.x;
    long total = (long)nB * M * C;
    if (i >= total) return;
    int c = (int)(i % C); long t = i / C;
    int m = (int)(t % M); int b = (int)(t / M);
    const float* p = in + (((long)b * M + m) * k) * C + c;
    float v = p[0];
    for (int j = 1; j < k; ++j) v = fmaxf(v, p[(long)j * C]);
    float* o = out + (long)b * out_bs + (long)m * C + c;
    *o = accum ? (*o + v) : v;
}

// row layernorm; writes f32 (y) and/or f16 (y16) output
__global__ void k_layernorm(const float* __restrict__ x, float* __restrict__ y,
                            _Float16* __restrict__ y16,
                            const float* __restrict__ g, const float* __restrict__ b, int C)
{
    long row = blockIdx.x;
    const float* px = x + row * (long)C;
    __shared__ float red[256];
    int tid = threadIdx.x;
    float s = 0.f;
    for (int c = tid; c < C; c += 256) s += px[c];
    red[tid] = s; __syncthreads();
    for (int st = 128; st > 0; st >>= 1) { if (tid < st) red[tid] += red[tid+st]; __syncthreads(); }
    float mean = red[0] / C; __syncthreads();
    float v = 0.f;
    for (int c = tid; c < C; c += 256) { float d = px[c] - mean; v += d * d; }
    red[tid] = v; __syncthreads();
    for (int st = 128; st > 0; st >>= 1) { if (tid < st) red[tid] += red[tid+st]; __syncthreads(); }
    float rstd = rsqrtf(red[0] / C + 1e-5f);
    for (int c = tid; c < C; c += 256) {
        float o = (px[c] - mean) * rstd * g[c] + b[c];
        if (y)   y[row * (long)C + c] = o;
        if (y16) y16[row * (long)C + c] = (_Float16)o;
    }
}

// row softmax: f32 in, f16 out (packed attention operand)
__global__ void k_softmax(const float* __restrict__ x, _Float16* __restrict__ out, int L)
{
    long row = blockIdx.x;
    const float* p = x + row * (long)L;
    _Float16* o = out + row * (long)L;
    __shared__ float red[256];
    int tid = threadIdx.x;
    float mx = -3.4e38f;
    for (int c = tid; c < L; c += 256) mx = fmaxf(mx, p[c]);
    red[tid] = mx; __syncthreads();
    for (int st = 128; st > 0; st >>= 1) { if (tid < st) red[tid] = fmaxf(red[tid], red[tid+st]); __syncthreads(); }
    mx = red[0]; __syncthreads();
    float s = 0.f;
    for (int c = tid; c < L; c += 256) s += expf(p[c] - mx);
    red[tid] = s; __syncthreads();
    for (int st = 128; st > 0; st >>= 1) { if (tid < st) red[tid] += red[tid+st]; __syncthreads(); }
    float inv = 1.f / red[0];
    for (int c = tid; c < L; c += 256) o[c] = (_Float16)(expf(p[c] - mx) * inv);
}

__global__ void k_interp3(const float* __restrict__ unk, long unk_bs,
                          const float* __restrict__ known, long kn_bs,
                          const float* __restrict__ feats, long ft_bs,
                          float* __restrict__ out, int Nunk, int Mkn, int C)
{
    int b = blockIdx.y, n = blockIdx.x;
    __shared__ int si[3]; __shared__ float sw[3];
    if (threadIdx.x == 0) {
        const float* u = unk + (long)b * unk_bs + 3L * n;
        float ux = u[0], uy = u[1], uz = u[2];
        float d0 = 3e38f, d1 = 3e38f, d2 = 3e38f; int i0 = 0, i1 = 0, i2 = 0;
        const float* Pk = known + (long)b * kn_bs;
        for (int m = 0; m < Mkn; ++m) {
            float dx = Pk[3*m]-ux, dy = Pk[3*m+1]-uy, dz = Pk[3*m+2]-uz;
            float d = dx*dx + dy*dy + dz*dz;
            if      (d < d0) { d2=d1;i2=i1; d1=d0;i1=i0; d0=d;i0=m; }
            else if (d < d1) { d2=d1;i2=i1; d1=d;i1=m; }
            else if (d < d2) { d2=d;i2=m; }
        }
        float w0 = 1.f/(d0+1e-8f), w1 = 1.f/(d1+1e-8f), w2 = 1.f/(d2+1e-8f);
        float wsum = w0 + w1 + w2;
        si[0]=i0; si[1]=i1; si[2]=i2;
        sw[0]=w0/wsum; sw[1]=w1/wsum; sw[2]=w2/wsum;
    }
    __syncthreads();
    const float* F = feats + (long)b * ft_bs;
    float* o = out + ((long)b * Nunk + n) * C;
    for (int c = threadIdx.x; c < C; c += blockDim.x)
        o[c] = sw[0]*F[(long)si[0]*C+c] + sw[1]*F[(long)si[1]*C+c] + sw[2]*F[(long)si[2]*C+c];
}

// column-concat, writes K-padded f16 GEMM operand
__global__ void k_concat(const float* __restrict__ a, long a_bs, int Ca,
                         const float* __restrict__ b, long b_bs, int Cb,
                         _Float16* __restrict__ out, int M, int nB, int KP)
{
    long i = blockIdx.x * (long)blockDim.x + threadIdx.x;
    long total = (long)nB * M * (long)KP;
    if (i >= total) return;
    int c = (int)(i % KP); long t = i / KP;
    int m = (int)(t % M); int bb = (int)(t / M);
    float v = 0.f;
    if (c < Ca) v = a[(long)bb * a_bs + (long)m * Ca + c];
    else if (c < Ca + Cb) v = b[(long)bb * b_bs + (long)m * Cb + (c - Ca)];
    out[i] = (_Float16)v;
}

__global__ void k_transpose2d(const float* __restrict__ in, float* __restrict__ out,
                              long batch, int R, int C)
{
    long i = blockIdx.x * (long)blockDim.x + threadIdx.x;
    long total = batch * (long)R * C;
    if (i >= total) return;
    int c = (int)(i % C); long t = i / C;
    int r = (int)(t % R); long b = t / R;
    out[b * (long)R * C + (long)c * R + r] = in[i];
}

__global__ void k_perm_bl8(const float* __restrict__ in, float* __restrict__ out,
                           int Bn, int Ln)
{
    int i = blockIdx.x * blockDim.x + threadIdx.x;
    if (i >= Bn * Ln * 8) return;
    int o = i % 8; int t = i / 8;
    int l = t % Ln; int b = t / Ln;
    out[(b * 8 + o) * Ln + l] = in[i];
}

// ------------------------- host-side orchestration -------------------------

struct Bump { char* p; };
static inline float* balloc(Bump& w, long nfloats) {
    float* r = (float*)w.p;
    w.p += ((nfloats * sizeof(float) + 255) & ~255UL);
    return r;
}
static inline _Float16* halloc(Bump& w, long n) { return (_Float16*)balloc(w, (n + 1) / 2); }
static inline int* ialloc(Bump& w, long n) { return (int*)balloc(w, n); }
static inline dim3 g1d(long n) { return dim3((unsigned)((n + 255) / 256)); }
static inline int clampi(int v, int lo, int hi) { return v < lo ? lo : (v > hi ? hi : v); }

static inline void gemmh(hipStream_t st, const _Float16* A, long lda,
                         const _Float16* Bm, long ldb,
                         void* C, long ldc, int M, int N, int K, float alpha,
                         const float* g, const float* b, const float* rb,
                         int act, int accum, int outF16, int outTrans)
{
    dim3 grid((N + GBN - 1) / GBN, (M + GBM - 1) / GBM);
    k_gemm_wmma<<<grid, 256, 0, st>>>(A, lda, Bm, ldb, C, ldc, M, N, K, alpha,
                                      g, b, rb, act, accum, outF16, outTrans);
}
static inline void packh(hipStream_t st, const float* src, long ld, int trans,
                         _Float16* dst, long M, long K)
{ k_packh<<<g1d(M * K), 256, 0, st>>>(src, ld, trans, dst, M, K, K); }

struct P4DP { const float *Wd,*gd,*bd,*Wf,*gf,*bf,*W1,*g1,*b1,*W2,*g2,*b2; };
struct CBRP { const float *W,*g,*b; };
struct TLP  { const float *ln1g,*ln1b,*wqkv,*wo,*bo,*ln2g,*ln2b,*w1,*b1,*w2,*b2; };

static void run_p4dconv(hipStream_t st, Bump& ws,
    const float* xyz, const float* feats,
    int Bn, int Ln, int Nin, int Cin,
    const P4DP& p, int c0, int c1, int c2,
    float radius, int kk, int sstride, int tks, int tpad0,
    float* out_xyz, float* out_feats)
{
    const int Mout = Nin / sstride;
    const int tr = tks / 2;
    const long rows = (long)Bn * Mout * kk;
    const long xyz_bs = (long)Ln * Nin * 3;
    const long ft_bs  = (long)Ln * Nin * (long)Cin;

    _Float16* Wdh = halloc(ws, (long)c0*4);   packh(st, p.Wd, 4,   0, Wdh, c0, 4);
    _Float16* Wfh = halloc(ws, (long)c0*Cin); packh(st, p.Wf, Cin, 0, Wfh, c0, Cin);
    _Float16* W1h = halloc(ws, (long)c1*c0);  packh(st, p.W1, c0,  0, W1h, c1, c0);
    _Float16* W2h = halloc(ws, (long)c2*c1);  packh(st, p.W2, c1,  0, W2h, c2, c1);

    int*      fidx = ialloc(ws, (long)Bn * Mout);
    float*    dws  = balloc(ws, (long)Bn * Nin);
    int*      nidx = ialloc(ws, rows);
    _Float16* disp = halloc(ws, rows * 4);
    _Float16* gft  = halloc(ws, rows * Cin);
    float*    h0   = balloc(ws, rows * c0);
    float*    h0b  = balloc(ws, rows * c0);
    _Float16* h0h  = halloc(ws, rows * c0);
    _Float16* hmh  = halloc(ws, rows * c1);
    float*    h2   = balloc(ws, rows * c2);

    for (int o = 0; o < Ln; ++o) {
        int creal = clampi(o + tr - tpad0, 0, Ln - 1);
        const float* cx = xyz + (long)creal * Nin * 3;
        float* anc = out_xyz + (long)o * Mout * 3;
        long anc_bs = (long)Ln * Mout * 3;

        k_fps<<<Bn, 256, 0, st>>>(cx, xyz_bs, Nin, Mout, fidx, Mout, dws);
        k_gather_rows<<<g1d((long)Bn*Mout*3), 256, 0, st>>>(cx, xyz_bs, fidx, Mout,
                                                            anc, anc_bs, Mout, 3, Bn);
        for (int i = 0; i < tks; ++i) {
            int ri = clampi(o + i - tpad0, 0, Ln - 1);
            const float* fx  = xyz   + (long)ri * Nin * 3;
            const float* ffr = feats + (long)ri * Nin * (long)Cin;
            k_ballq<<<g1d((long)Bn*Mout), 256, 0, st>>>(fx, xyz_bs, anc, anc_bs,
                                                        nidx, Nin, Mout, kk, radius*radius, Bn);
            k_group_disp<<<g1d(rows), 256, 0, st>>>(fx, xyz_bs, anc, anc_bs, nidx,
                                                    disp, Mout, kk, (float)(i - tr), Bn);
            k_gather_rows_h<<<g1d(rows*Cin), 256, 0, st>>>(ffr, ft_bs, nidx, (long)Mout*kk,
                                                           gft, (long)Mout*kk*Cin, Mout*kk, Cin, Bn);
            gemmh(st, disp, 4, Wdh, 4, h0, c0, (int)rows, c0, 4, 1.f, p.gd, p.bd, 0, 1, 0, 0, 0);
            gemmh(st, gft, Cin, Wfh, Cin, h0b, c0, (int)rows, c0, Cin, 1.f, p.gf, p.bf, 0, 1, 0, 0, 0);
            k_addh<<<g1d(rows*c0), 256, 0, st>>>(h0, h0b, h0h, rows * c0);
            gemmh(st, h0h, c0, W1h, c0, hmh, c1, (int)rows, c1, c0, 1.f, p.g1, p.b1, 0, 1, 0, 1, 0);
            gemmh(st, hmh, c1, W2h, c1, h2, c2, (int)rows, c2, c1, 1.f, p.g2, p.b2, 0, 1, 0, 0, 0);
            k_maxpool<<<g1d((long)Bn*Mout*c2), 256, 0, st>>>(h2,
                out_feats + (long)o * Mout * c2, (long)Ln * Mout * (long)c2,
                Mout, kk, c2, Bn, i > 0 ? 1 : 0);
        }
    }
}

static void run_deconv(hipStream_t st, Bump& ws,
    const float* unk, int Nunk,
    const float* kn, int Mkn,
    const float* feats, int Cf, int featsFrameMajor,
    const float* skip, int Cs,
    const CBRP* cbr, int c1, int c2,
    float* out,                       // frame-major (L,B,Nunk,c2)
    int Bn, int Ln)
{
    const int Ct = Cf + Cs;
    const int KP = (Ct + 7) & ~7;
    _Float16* W0h = halloc(ws, (long)c1*KP);
    k_packh<<<g1d((long)c1*KP), 256, 0, st>>>(cbr[0].W, Ct, 0, W0h, c1, Ct, KP);
    _Float16* W1h = halloc(ws, (long)c2*c1); packh(st, cbr[1].W, c1, 0, W1h, c2, c1);

    float*    ip  = balloc(ws, (long)Bn * Nunk * Cf);
    _Float16* cc  = halloc(ws, (long)Bn * Nunk * KP);
    _Float16* h1h = halloc(ws, (long)Bn * Nunk * c1);
    for (int t = 0; t < Ln; ++t) {
        const float* fbase; long fbs;
        if (featsFrameMajor) { fbase = feats + (long)t * Bn * Mkn * Cf; fbs = (long)Mkn * Cf; }
        else                 { fbase = feats + (long)t * Mkn * Cf;      fbs = (long)Ln * Mkn * Cf; }
        k_interp3<<<dim3(Nunk, Bn), 128, 0, st>>>(
            unk + (long)t * Nunk * 3, (long)Ln * Nunk * 3,
            kn  + (long)t * Mkn  * 3, (long)Ln * Mkn  * 3,
            fbase, fbs, ip, Nunk, Mkn, Cf);
        k_concat<<<g1d((long)Bn*Nunk*KP), 256, 0, st>>>(
            ip, (long)Nunk * Cf, Cf,
            skip + (long)t * Nunk * Cs, (long)Ln * Nunk * (long)Cs, Cs,
            cc, Nunk, Bn, KP);
        gemmh(st, cc, KP, W0h, KP, h1h, c1, Bn*Nunk, c1, KP,
              1.f, cbr[0].g, cbr[0].b, 0, 1, 0, 1, 0);
        gemmh(st, h1h, c1, W1h, c1, out + (long)t * Bn * Nunk * c2, c2,
              Bn*Nunk, c2, c1, 1.f, cbr[1].g, cbr[1].b, 0, 1, 0, 0, 0);
    }
}

extern "C" void kernel_launch(void* const* d_in, const int* in_sizes, int n_in,
                              void* d_out, int out_size, void* d_ws, size_t ws_size,
                              hipStream_t stream)
{
    const int B = 2, L = 3, N = 16384, KNB = 9, NT = 2048;
    const int M1 = 4096, M2 = 1024, M3 = 256, M4 = 128;
    const float R = 0.9f;

    const float* xyzs = (const float*)d_in[0];
    const float* rgbs = (const float*)d_in[1];
    const float* tmpl = (const float*)d_in[2];

    int cur = 3;
    auto nx = [&]() { return (const float*)d_in[cur++]; };
    auto rd_p4d = [&](P4DP& p) {
        p.Wd=nx(); p.gd=nx(); p.bd=nx(); p.Wf=nx(); p.gf=nx(); p.bf=nx();
        p.W1=nx(); p.g1=nx(); p.b1=nx(); p.W2=nx(); p.g2=nx(); p.b2=nx();
    };
    P4DP conv1, conv2, conv3, conv4, tconv;
    rd_p4d(conv1); rd_p4d(conv2); rd_p4d(conv3); rd_p4d(conv4);
    TLP tl[2];
    for (int i = 0; i < 2; ++i) {
        tl[i].ln1g=nx(); tl[i].ln1b=nx(); tl[i].wqkv=nx(); tl[i].wo=nx(); tl[i].bo=nx();
        tl[i].ln2g=nx(); tl[i].ln2b=nx(); tl[i].w1=nx();  tl[i].b1=nx();
        tl[i].w2=nx();   tl[i].b2=nx();
    }
    rd_p4d(tconv);
    CBRP dc4[2], dc3[2], dc2[2], dc1[2];
    auto rd_cbr2 = [&](CBRP* c) {
        c[0].W=nx(); c[0].g=nx(); c[0].b=nx();
        c[1].W=nx(); c[1].g=nx(); c[1].b=nx();
    };
    rd_cbr2(dc4); rd_cbr2(dc3); rd_cbr2(dc2); rd_cbr2(dc1);
    const float *outW=nx(), *outB=nx();
    const float *cn1W=nx(), *cn1B=nx();
    const float *cn2W=nx(), *cn2B=nx();
    const float *toW=nx(),  *toB=nx();
    const float *to2W=nx(), *to2B=nx();
    const float *mlnG=nx(), *mlnB=nx();
    const float *mw1=nx(),  *mb1=nx();
    const float *mw2=nx(),  *mb2=nx();

    Bump ws{(char*)d_ws};

    float* f0 = balloc(ws, (long)B*L*N*3);
    k_transpose2d<<<g1d((long)B*L*3*N), 256, 0, stream>>>(rgbs, f0, (long)B*L, 3, N);

    float* x1 = balloc(ws, (long)B*L*M1*3);  float* f1 = balloc(ws, (long)B*L*M1*128);
    float* x2 = balloc(ws, (long)B*L*M2*3);  float* f2 = balloc(ws, (long)B*L*M2*256);
    float* x3 = balloc(ws, (long)B*L*M3*3);  float* f3 = balloc(ws, (long)B*L*M3*512);
    float* x4 = balloc(ws, (long)B*L*M4*3);  float* f4 = balloc(ws, (long)B*L*M4*1024);
    run_p4dconv(stream, ws, xyzs, f0, B, L, N,  3,   conv1, 32, 64, 128,  R,     KNB, 4, 1, 0, x1, f1);
    run_p4dconv(stream, ws, x1,   f1, B, L, M1, 128, conv2, 128,128,256,  2.f*R, KNB, 4, 1, 0, x2, f2);
    run_p4dconv(stream, ws, x2,   f2, B, L, M2, 256, conv3, 256,256,512,  4.f*R, KNB, 4, 3, 1, x3, f3);
    run_p4dconv(stream, ws, x3,   f3, B, L, M3, 512, conv4, 512,512,1024, 8.f*R, KNB, 2, 1, 0, x4, f4);

    // ---------------- transformer (all-f16 WMMA path) ----------------
    const int S = L * M4, D = 1024, H = 4, DH = 256;
    const long ROWS = (long)B * S;
    float* x = f4;
    k_relu<<<g1d(ROWS * D), 256, 0, stream>>>(x, ROWS * D);

    _Float16 *wqkvh[2], *woh[2], *w1h[2], *w2h[2];
    for (int i = 0; i < 2; ++i) {
        wqkvh[i] = halloc(ws, (long)3*D*D); packh(stream, tl[i].wqkv, D, 0, wqkvh[i], 3*D, D);
        woh[i]   = halloc(ws, (long)D*D);   packh(stream, tl[i].wo,   D, 0, woh[i],   D, D);
        w1h[i]   = halloc(ws, (long)D*D);   packh(stream, tl[i].w1,   D, 0, w1h[i],   D, D);
        w2h[i]   = halloc(ws, (long)D*D);   packh(stream, tl[i].w2,   D, 0, w2h[i],   D, D);
    }
    float*    sc   = balloc(ws, (long)B * H * S * S);
    _Float16* ph   = halloc(ws, ROWS * D);
    _Float16* qkvh = halloc(ws, ROWS * 3 * D);
    _Float16* vTh  = halloc(ws, (long)B * H * DH * S);
    _Float16* atth = halloc(ws, (long)B * H * S * S);
    _Float16* oh   = halloc(ws, ROWS * D);
    _Float16* m1h  = halloc(ws, ROWS * D);

    for (int ly = 0; ly < 2; ++ly) {
        k_layernorm<<<(unsigned)ROWS, 256, 0, stream>>>(x, (float*)0, ph, tl[ly].ln1g, tl[ly].ln1b, D);
        gemmh(stream, ph, D, wqkvh[ly], D, qkvh, 3*D, (int)ROWS, 3*D, D, 1.f, 0, 0, 0, 0, 0, 1, 0);
        for (int b = 0; b < B; ++b)
            for (int h = 0; h < H; ++h) {
                long bh = b*H + h;
                k_packh16<<<g1d((long)DH*S), 256, 0, stream>>>(
                    qkvh + (long)b*S*3*D + 2*D + h*DH, 3*D, 1, vTh + bh*DH*S, DH, S);
                const _Float16* q  = qkvh + (long)b*S*3*D + h*DH;
                const _Float16* kp = qkvh + (long)b*S*3*D + D + h*DH;
                gemmh(stream, q, 3*D, kp, 3*D, sc + bh*S*S, S, S, S, DH,
                      0.0625f, 0, 0, 0, 0, 0, 0, 0);
            }
        k_softmax<<<(unsigned)((long)B*H*S), 256, 0, stream>>>(sc, atth, S);
        for (int b = 0; b < B; ++b)
            for (int h = 0; h < H; ++h) {
                long bh = b*H + h;
                gemmh(stream, atth + bh*S*S, S, vTh + bh*DH*S, S,
                      oh + (long)b*S*D + h*DH, D, S, DH, S, 1.f, 0, 0, 0, 0, 0, 1, 0);
            }
        gemmh(stream, oh, D, woh[ly], D, x, D, (int)ROWS, D, D, 1.f, 0, tl[ly].bo, 0, 0, 1, 0, 0);
        k_layernorm<<<(unsigned)ROWS, 256, 0, stream>>>(x, (float*)0, ph, tl[ly].ln2g, tl[ly].ln2b, D);
        gemmh(stream, ph, D, w1h[ly], D, m1h, D, (int)ROWS, D, D, 1.f, 0, tl[ly].b1, 0, 2, 0, 1, 0);
        gemmh(stream, m1h, D, w2h[ly], D, x, D, (int)ROWS, D, D, 1.f, 0, tl[ly].b2, 0, 0, 1, 0, 0);
    }
    float* f4t = x;

    // ---------------- deconv chain ----------------
    float* fd4 = balloc(ws, (long)L*B*M3*256);
    float* fd3 = balloc(ws, (long)L*B*M2*256);
    float* fd2 = balloc(ws, (long)L*B*M1*128);
    float* fd1 = balloc(ws, (long)L*B*N*128);
    run_deconv(stream, ws, x3, M3, x4, M4, f4t, 1024, 0, f3, 512, dc4, 256, 256, fd4, B, L);
    run_deconv(stream, ws, x2, M2, x3, M3, fd4, 256,  1, f2, 256, dc3, 256, 256, fd3, B, L);
    run_deconv(stream, ws, x1, M1, x2, M2, fd3, 256,  1, f1, 128, dc2, 128, 128, fd2, B, L);
    run_deconv(stream, ws, xyzs, N, x1, M1, fd2, 128, 1, f0, 3,   dc1, 128, 128, fd1, B, L);

    // ---------------- heads ----------------
    float* out0 = (float*)d_out;
    _Float16* outWh = halloc(ws, 2*128);
    packh(stream, outW, 128, 0, outWh, 2, 128);
    _Float16* fd1h = halloc(ws, (long)L*B*N*128);
    packh(stream, fd1, 128, 0, fd1h, (long)L*B*N, 128);
    for (int b = 0; b < B; ++b)
        for (int t = 0; t < L; ++t)
            gemmh(stream, outWh, 128, fd1h + ((long)t*B + b)*N*128, 128,
                  out0 + ((long)b*L + t)*2*N, N, 2, N, 128, 1.f, 0, 0, outB, 0, 0, 0, 0);

    // template branch
    float* tfz = balloc(ws, (long)NT * 3);
    k_fill<<<g1d((long)NT*3), 256, 0, stream>>>(tfz, (long)NT*3, 0.f);
    float* txA = balloc(ws, (long)512 * 3);
    float* tft = balloc(ws, (long)512 * 128);
    run_p4dconv(stream, ws, tmpl, tfz, 1, 1, NT, 3, tconv, 32, 64, 128, R, KNB, 4, 1, 0, txA, tft);

    // tci[b,l] (256,512): rows 0..127 = nfd (WMMA), rows 128..255 = tfeat^T
    _Float16* to2Wh = halloc(ws, (long)512 * N);
    packh(stream, to2W, N, 0, to2Wh, 512, N);
    _Float16* fdTh = halloc(ws, (long)128 * N);
    float* tci = balloc(ws, (long)B*L*256*512);
    for (int b = 0; b < B; ++b)
        for (int t = 0; t < L; ++t) {
            const float* fbl = fd1 + ((long)t*B + b) * N * 128;
            float* tbl = tci + ((long)b*L + t) * 256 * 512;
            packh(stream, fbl, 128, 1, fdTh, 128, N);
            gemmh(stream, fdTh, N, to2Wh, N, tbl, 512, 128, 512, N, 1.f, 0, to2B, 0, 0, 0, 0, 0);
            k_transpose2d<<<g1d((long)512*128), 256, 0, stream>>>(tft, tbl + 128L*512, 1, 512, 128);
        }

    // tmo[b,l] (8,512) = toW(8,256) x tci^T + toB (row bias)
    _Float16* toWh = halloc(ws, 8*256);
    packh(stream, toW, 256, 0, toWh, 8, 256);
    _Float16* tciT = halloc(ws, (long)512*256);
    float* tmo = balloc(ws, (long)B*L*8*512);
    for (int bl = 0; bl < B*L; ++bl) {
        packh(stream, tci + (long)bl*256*512, 512, 1, tciT, 512, 256);
        gemmh(stream, toWh, 256, tciT, 256, tmo + (long)bl*8*512, 512,
              8, 512, 256, 1.f, 0, 0, toB, 0, 0, 0, 0);
    }

    // mlp head -> template_out (B,8,L)
    _Float16* mw1h = halloc(ws, 64*512); packh(stream, mw1, 512, 0, mw1h, 64, 512);
    _Float16* mw2h = halloc(ws, 64);     packh(stream, mw2, 64, 0, mw2h, 1, 64);
    _Float16* mlnh = halloc(ws, (long)B*L*8*512);
    _Float16* mhh  = halloc(ws, (long)B*L*8*64);
    float* mo = balloc(ws, (long)B*L*8);
    k_layernorm<<<(unsigned)(B*L*8), 256, 0, stream>>>(tmo, (float*)0, mlnh, mlnG, mlnB, 512);
    gemmh(stream, mlnh, 512, mw1h, 512, mhh, 64, B*L*8, 64, 512, 1.f, 0, mb1, 0, 2, 0, 1, 0);
    gemmh(stream, mhh, 64, mw2h, 64, mo, 1, B*L*8, 1, 64, 1.f, 0, mb2, 0, 0, 0, 0, 0);
    float* out1 = out0 + (long)B*L*2*N;
    k_perm_bl8<<<g1d((long)B*L*8), 256, 0, stream>>>(mo, out1, B, L);

    // c1[b,l] = tci x cn1W^T + cn1B, stored transposed as f16 (16384 x 256),
    // then final_out[b,l] (2,16384) = cn2W(2,256) x c1^T^T + cn2B (row bias)
    _Float16* cn1Wh = halloc(ws, (long)N * 512);
    packh(stream, cn1W, 512, 0, cn1Wh, N, 512);
    _Float16* cn2Wh = halloc(ws, 2*256);
    packh(stream, cn2W, 256, 0, cn2Wh, 2, 256);
    _Float16* tcih = halloc(ws, (long)256 * 512);
    _Float16* c1T  = halloc(ws, (long)N * 256);
    float* out2 = out1 + (long)B*8*L;
    for (int bl = 0; bl < B*L; ++bl) {
        packh(stream, tci + (long)bl*256*512, 512, 0, tcih, 256, 512);
        gemmh(stream, tcih, 512, cn1Wh, 512, c1T, 256, 256, N, 512,
              1.f, 0, cn1B, 0, 0, 0, 1, 1 /*f16, transposed store*/);
        gemmh(stream, cn2Wh, 256, c1T, 256, out2 + (long)bl*2*N, N,
              2, N, 256, 1.f, 0, 0, cn2B, 0, 0, 0, 0);
    }
}
